// SwinTransformerLayer_7550552506773
// MI455X (gfx1250) — matrix operand order repaired and checked by reading, not compiled, not run on hardware
//
#include <hip/hip_runtime.h>
#include <hip/hip_bf16.h>
#include <cmath>

// ---------------------------------------------------------------------------
// Swin Transformer layer (B=32, H=W=56, C=384, heads=12, hd=32, WS=7) for
// gfx1250. All GEMMs run on v_wmma_f32_16x16x32_f16 (f16 in, f32 accum).
// GEMM waves compute 64x64 output tiles (16 WMMAs per 16 b128 loads per
// K-step). Attention pads N=49->64 via zero-filled LDS staging.
// ---------------------------------------------------------------------------

typedef __attribute__((ext_vector_type(16))) _Float16 v16h;
typedef __attribute__((ext_vector_type(8)))  _Float16 v8h;
typedef __attribute__((ext_vector_type(8)))  float    v8f;

#define kWS      7
#define kN       49                 // tokens per window
#define kC       384
#define kHeads   12
#define kHd      32
#define kB       32
#define kH       56
#define kW       56
#define kL       (kH * kW)          // 3136
#define kNW      ((kH / kWS) * (kW / kWS))   // 64 windows per image
#define kWindows (kB * kNW)         // 2048
#define kM       (kWindows * kN)    // 100352 rows (== kB*kL)
#define k3C      (3 * kC)           // 1152
#define kHidden  (4 * kC)           // 1536
#define kShift   (kWS / 2)          // 3
#define kQScale  0.17677669529663688f   // 1/sqrt(32)

// -------------------------------- helpers ----------------------------------

__device__ inline v8f zero8() {
  v8f z;
#pragma unroll
  for (int i = 0; i < 8; ++i) z[i] = 0.0f;
  return z;
}

__device__ inline v8f wmma_f16(v16h a, v16h b, v8f c) {
  // D = A(16x32) * B(32x16) + C, f32 accumulate
  return __builtin_amdgcn_wmma_f32_16x16x32_f16(
      /*neg_a=*/false, a, /*neg_b=*/false, b,
      /*c_mod=*/(short)0, c, /*reuse_a=*/false, /*reuse_b=*/false);
}

// A fragment: 16x32 (MxK) from row-major src (ld halves per row).
// ISA 7.12.2 layout: lane m = lane&15; lower lanes hold K={kb..kb+7,
// 16+kb..16+kb+7} with kb=0, upper lanes kb=8. Two aligned 128-bit loads.
__device__ inline v16h load_a_f16(const _Float16* __restrict__ src, int ld,
                                  int row0, int k0) {
  const int lane = threadIdx.x & 31;
  const int m    = lane & 15;
  const int kb   = (lane >> 4) ? 8 : 0;
  const _Float16* p = src + (size_t)(row0 + m) * ld + k0 + kb;
  v8h lo = *(const v8h*)(p);        // K = kb .. kb+7
  v8h hi = *(const v8h*)(p + 16);   // K = 16+kb .. 16+kb+7
  return __builtin_shufflevector(lo, hi, 0, 1, 2, 3, 4, 5, 6, 7,
                                         8, 9, 10, 11, 12, 13, 14, 15);
}

// B fragment: 32x16 (KxN) where B[k][n] = src[col0+n][k] (src row-major
// [Ncols, K], i.e. weight kept in natural [out,in] orientation, or an
// explicitly transposed tile). Lane n = lane&15; lower lanes K={0..7,16..23},
// upper lanes K={8..15,24..31} (same order as the A fragment).
__device__ inline v16h load_bT_f16(const _Float16* __restrict__ src, int ld,
                                   int col0, int k0) {
  const int lane = threadIdx.x & 31;
  const int n    = lane & 15;
  const int kb   = (lane >> 4) ? 8 : 0;
  const _Float16* p = src + (size_t)(col0 + n) * ld + k0 + kb;
  v8h lo = *(const v8h*)(p);        // K = kb .. kb+7
  v8h hi = *(const v8h*)(p + 16);   // K = 16+kb .. 16+kb+7
  return __builtin_shufflevector(lo, hi, 0, 1, 2, 3, 4, 5, 6, 7,
                                         8, 9, 10, 11, 12, 13, 14, 15);
}

// 64x64 output block per wave: 4x4 tiles of 16x16, K loop in steps of 32.
// Per K-step: 4 A-frag loads + 4 B-frag loads (16 x b128) feed 16 WMMAs.
__device__ inline void gemm64x64(const _Float16* __restrict__ A, int lda, int row0,
                                 const _Float16* __restrict__ Wt, int ldw, int col0,
                                 int kdim, v8f (&acc)[4][4]) {
#pragma unroll
  for (int i = 0; i < 4; ++i)
#pragma unroll
    for (int j = 0; j < 4; ++j) acc[i][j] = zero8();

  for (int k0 = 0; k0 < kdim; k0 += 32) {
    v16h a[4];
#pragma unroll
    for (int i = 0; i < 4; ++i) a[i] = load_a_f16(A, lda, row0 + 16 * i, k0);
#pragma unroll
    for (int j = 0; j < 4; ++j) {
      v16h b = load_bT_f16(Wt, ldw, col0 + 16 * j, k0);
#pragma unroll
      for (int i = 0; i < 4; ++i) acc[i][j] = wmma_f16(a[i], b, acc[i][j]);
    }
  }
}

// ------------------------------- kernels -----------------------------------

__global__ void k_f32_to_f16(const float* __restrict__ in,
                             _Float16* __restrict__ out, int n) {
  int i = blockIdx.x * 256 + threadIdx.x;
  if (i < n) out[i] = (_Float16)in[i];
}

// LN1 + roll(-3,-3) + window partition -> f16 rows [kM, kC] in window order.
__global__ __launch_bounds__(256) void k_ln_window(
    const float* __restrict__ x, const float* __restrict__ w,
    const float* __restrict__ b, _Float16* __restrict__ out) {
  const int wv = threadIdx.x >> 5, lane = threadIdx.x & 31;
  const int row = blockIdx.x * 8 + wv;
  const int win = row / kN, nt = row % kN;
  const int bb = win / kNW, wrem = win % kNW;
  const int i = (wrem / (kW / kWS)) * kWS + nt / kWS;
  const int j = (wrem % (kW / kWS)) * kWS + nt % kWS;
  const int si = (i + kShift) % kH, sj = (j + kShift) % kW;
  const float* xr = x + ((size_t)bb * kL + si * kW + sj) * kC;

  float vals[12];
  float s = 0.0f;
#pragma unroll
  for (int t = 0; t < 12; ++t) { vals[t] = xr[lane + t * 32]; s += vals[t]; }
#pragma unroll
  for (int m = 16; m > 0; m >>= 1) s += __shfl_xor(s, m, 32);
  const float mean = s * (1.0f / (float)kC);
  float vsum = 0.0f;
#pragma unroll
  for (int t = 0; t < 12; ++t) { float d = vals[t] - mean; vsum += d * d; }
#pragma unroll
  for (int m = 16; m > 0; m >>= 1) vsum += __shfl_xor(vsum, m, 32);
  const float inv = rsqrtf(vsum * (1.0f / (float)kC) + 1e-5f);
  _Float16* orow = out + (size_t)row * kC;
#pragma unroll
  for (int t = 0; t < 12; ++t) {
    const int c = lane + t * 32;
    orow[c] = (_Float16)((vals[t] - mean) * inv * w[c] + b[c]);
  }
}

// LN2 over xbuf rows (token order) -> f16 [kM, kC].
__global__ __launch_bounds__(256) void k_ln_plain(
    const float* __restrict__ x, const float* __restrict__ w,
    const float* __restrict__ b, _Float16* __restrict__ out) {
  const int wv = threadIdx.x >> 5, lane = threadIdx.x & 31;
  const int row = blockIdx.x * 8 + wv;
  const float* xr = x + (size_t)row * kC;
  float vals[12];
  float s = 0.0f;
#pragma unroll
  for (int t = 0; t < 12; ++t) { vals[t] = xr[lane + t * 32]; s += vals[t]; }
#pragma unroll
  for (int m = 16; m > 0; m >>= 1) s += __shfl_xor(s, m, 32);
  const float mean = s * (1.0f / (float)kC);
  float vsum = 0.0f;
#pragma unroll
  for (int t = 0; t < 12; ++t) { float d = vals[t] - mean; vsum += d * d; }
#pragma unroll
  for (int m = 16; m > 0; m >>= 1) vsum += __shfl_xor(vsum, m, 32);
  const float inv = rsqrtf(vsum * (1.0f / (float)kC) + 1e-5f);
  _Float16* orow = out + (size_t)row * kC;
#pragma unroll
  for (int t = 0; t < 12; ++t) {
    const int c = lane + t * 32;
    orow[c] = (_Float16)((vals[t] - mean) * inv * w[c] + b[c]);
  }
}

// QKV GEMM: [kM,384] x [384,1152]^T; epilogue: +bias, scale q by 1/sqrt(hd),
// scatter to q/k/v buffers laid out [win][head][n][d].
__global__ __launch_bounds__(32) void k_gemm_qkv(
    const _Float16* __restrict__ xw, const _Float16* __restrict__ wq,
    const float* __restrict__ bias,
    _Float16* __restrict__ qb, _Float16* __restrict__ kb,
    _Float16* __restrict__ vb) {
  const int col0 = blockIdx.x * 64;
  const int row0 = blockIdx.y * 64;
  v8f acc[4][4];
  gemm64x64(xw, kC, row0, wq, kC, col0, kC, acc);
  const int lane = threadIdx.x & 31;
  const int n = lane & 15, mb = (lane >> 4) * 8;
#pragma unroll
  for (int ti = 0; ti < 4; ++ti)
#pragma unroll
    for (int tj = 0; tj < 4; ++tj)
#pragma unroll
      for (int i = 0; i < 8; ++i) {
        const int r = row0 + ti * 16 + mb + i;
        const int c = col0 + tj * 16 + n;
        float v = acc[ti][tj][i] + bias[c];
        const int which = c / kC;
        const int head  = (c % kC) / kHd;
        const int d     = c % kHd;
        const int win = r / kN, tok = r % kN;
        const size_t idx = (((size_t)win * kHeads + head) * kN + tok) * kHd + d;
        if (which == 0)      qb[idx] = (_Float16)(v * kQScale);
        else if (which == 1) kb[idx] = (_Float16)v;
        else                 vb[idx] = (_Float16)v;
      }
}

// Windowed attention: one wave per (window, head). N=49 padded to 64 via
// zero-filled LDS staging; all WMMA fragments come from LDS (branch-free).
__global__ __launch_bounds__(32) void k_attn(
    const _Float16* __restrict__ qb, const _Float16* __restrict__ kb,
    const _Float16* __restrict__ vb, const float* __restrict__ bias_table,
    const float* __restrict__ mask, _Float16* __restrict__ obuf) {
  __shared__ _Float16 Qs[64 * 32];  // rows 49..63 zero
  __shared__ _Float16 Ks[64 * 32];  // rows 49..63 zero
  __shared__ _Float16 Vt[32 * 64];  // transposed: Vt[d][m], cols 49..63 zero
  __shared__ float    S[64 * 65];   // scores (padded stride)
  __shared__ _Float16 P[64 * 64];   // softmax probs (f16 for WMMA reload)

  const int wh  = blockIdx.x;
  const int win = wh / kHeads, head = wh % kHeads;
  const size_t base = ((size_t)win * kHeads + head) * kN * kHd;
  const int lane = threadIdx.x & 31;
  const int n = lane & 15, mb = (lane >> 4) * 8;
  const int mwin = win % kNW;

  // ---- stage Q, K (row-major, zero padded) and V^T into LDS --------------
  {
    const unsigned* q32 = (const unsigned*)(qb + base);
    const unsigned* k32 = (const unsigned*)(kb + base);
    const unsigned* v32 = (const unsigned*)(vb + base);
    unsigned* qs = (unsigned*)Qs;
    unsigned* ks = (unsigned*)Ks;
    unsigned short* vt = (unsigned short*)Vt;
    // 64 rows * 32 halves = 1024 dwords; valid region = 49*16 = 784 dwords
    for (int i = lane; i < 1024; i += 32) {
      const bool ok = i < (kN * kHd / 2);
      qs[i] = ok ? q32[i] : 0u;
      ks[i] = ok ? k32[i] : 0u;
      const unsigned wv = ok ? v32[i] : 0u;
      const int m = i >> 4;            // source row (token)
      const int d = (i & 15) * 2;      // source column pair (head dim)
      vt[(d + 0) * 64 + m] = (unsigned short)(wv & 0xffffu);
      vt[(d + 1) * 64 + m] = (unsigned short)(wv >> 16);
    }
  }
  __syncthreads();

  // ---- S = Q*K^T + rel_bias + mask ---------------------------------------
  for (int mt = 0; mt < 4; ++mt) {
    v16h aq = load_a_f16(Qs, kHd, mt * 16, 0);
    for (int nt = 0; nt < 4; ++nt) {
      v16h bk = load_bT_f16(Ks, kHd, nt * 16, 0);   // B[k][n] = K[n][k]
      v8f acc = zero8();
      acc = wmma_f16(aq, bk, acc);
#pragma unroll
      for (int i = 0; i < 8; ++i) {
        const int r = mt * 16 + mb + i;
        const int c = nt * 16 + n;
        float v = acc[i];
        if (r < kN && c < kN) {
          const int i1 = r / kWS, j1 = r % kWS, i2 = c / kWS, j2 = c % kWS;
          const int bidx = (i1 - i2 + kWS - 1) * (2 * kWS - 1) + (j1 - j2 + kWS - 1);
          v += bias_table[(size_t)bidx * kHeads + head];
          v += mask[((size_t)mwin * kN + r) * kN + c];
        }
        S[r * 65 + c] = v;
      }
    }
  }
  __syncthreads();

  // ---- row softmax (rows lane and lane+32) -------------------------------
  for (int rr = 0; rr < 2; ++rr) {
    const int r = lane + rr * 32;
    float mx = -1e30f;
    for (int m = 0; m < kN; ++m) mx = fmaxf(mx, S[r * 65 + m]);
    float sum = 0.0f;
    for (int m = 0; m < kN; ++m) {
      const float e = __expf(S[r * 65 + m] - mx);
      sum += e;
      S[r * 65 + m] = e;
    }
    const float inv = 1.0f / sum;
    for (int m = 0; m < 64; ++m)
      P[r * 64 + m] = (m < kN) ? (_Float16)(S[r * 65 + m] * inv) : (_Float16)0.0f;
  }
  __syncthreads();

  // ---- O = P * V  (B[k][n] = V[k][n] = Vt[n][k] -> contiguous loads) -----
  v16h bv[2][2];
#pragma unroll
  for (int ks2 = 0; ks2 < 2; ++ks2)
#pragma unroll
    for (int nt = 0; nt < 2; ++nt)
      bv[ks2][nt] = load_bT_f16(Vt, 64, nt * 16, ks2 * 32);

  for (int mt = 0; mt < 4; ++mt) {
    v16h a0 = load_a_f16(P, 64, mt * 16, 0);
    v16h a1 = load_a_f16(P, 64, mt * 16, 32);
    for (int nt = 0; nt < 2; ++nt) {
      v8f acc = zero8();
      acc = wmma_f16(a0, bv[0][nt], acc);
      acc = wmma_f16(a1, bv[1][nt], acc);
#pragma unroll
      for (int i = 0; i < 8; ++i) {
        const int r = mt * 16 + mb + i;
        if (r < kN)
          obuf[((size_t)win * kN + r) * kC + head * kHd + nt * 16 + n] =
              (_Float16)acc[i];
      }
    }
  }
}

// proj GEMM + bias + window-reverse + roll(+3,+3) + residual -> xbuf (f32).
__global__ __launch_bounds__(32) void k_gemm_proj(
    const _Float16* __restrict__ obuf, const _Float16* __restrict__ wp,
    const float* __restrict__ bias, const float* __restrict__ x_in,
    float* __restrict__ xbuf) {
  const int col0 = blockIdx.x * 64;
  const int row0 = blockIdx.y * 64;
  v8f acc[4][4];
  gemm64x64(obuf, kC, row0, wp, kC, col0, kC, acc);
  const int lane = threadIdx.x & 31;
  const int n = lane & 15, mb = (lane >> 4) * 8;
#pragma unroll
  for (int ti = 0; ti < 4; ++ti)
#pragma unroll
    for (int tj = 0; tj < 4; ++tj)
#pragma unroll
      for (int i = 0; i < 8; ++i) {
        const int r = row0 + ti * 16 + mb + i;
        const int c = col0 + tj * 16 + n;
        const float v = acc[ti][tj][i] + bias[c];
        const int win = r / kN, tok = r % kN;
        const int bb = win / kNW, wrem = win % kNW;
        const int ii = (wrem / (kW / kWS)) * kWS + tok / kWS;
        const int jj = (wrem % (kW / kWS)) * kWS + tok % kWS;
        const int oi = (ii + kShift) % kH, oj = (jj + kShift) % kW;
        const size_t idx = ((size_t)bb * kL + oi * kW + oj) * kC + c;
        xbuf[idx] = x_in[idx] + v;
      }
}

// MLP fc1 GEMM + bias + exact GELU -> f16 hidden [kM, kHidden].
__global__ __launch_bounds__(32) void k_gemm_mlp1(
    const _Float16* __restrict__ h16, const _Float16* __restrict__ w1,
    const float* __restrict__ bias, _Float16* __restrict__ hid) {
  const int col0 = blockIdx.x * 64;
  const int row0 = blockIdx.y * 64;
  v8f acc[4][4];
  gemm64x64(h16, kC, row0, w1, kC, col0, kC, acc);
  const int lane = threadIdx.x & 31;
  const int n = lane & 15, mb = (lane >> 4) * 8;
#pragma unroll
  for (int ti = 0; ti < 4; ++ti)
#pragma unroll
    for (int tj = 0; tj < 4; ++tj)
#pragma unroll
      for (int i = 0; i < 8; ++i) {
        const int r = row0 + ti * 16 + mb + i;
        const int c = col0 + tj * 16 + n;
        float v = acc[ti][tj][i] + bias[c];
        v = 0.5f * v * (1.0f + erff(v * 0.7071067811865476f));
        hid[(size_t)r * kHidden + c] = (_Float16)v;
      }
}

// MLP fc2 GEMM + bias + residual(xbuf) -> final output (f32).
__global__ __launch_bounds__(32) void k_gemm_mlp2(
    const _Float16* __restrict__ hid, const _Float16* __restrict__ w2,
    const float* __restrict__ bias, const float* __restrict__ xbuf,
    float* __restrict__ out) {
  const int col0 = blockIdx.x * 64;
  const int row0 = blockIdx.y * 64;
  v8f acc[4][4];
  gemm64x64(hid, kHidden, row0, w2, kHidden, col0, kHidden, acc);
  const int lane = threadIdx.x & 31;
  const int n = lane & 15, mb = (lane >> 4) * 8;
#pragma unroll
  for (int ti = 0; ti < 4; ++ti)
#pragma unroll
    for (int tj = 0; tj < 4; ++tj)
#pragma unroll
      for (int i = 0; i < 8; ++i) {
        const int r = row0 + ti * 16 + mb + i;
        const int c = col0 + tj * 16 + n;
        const size_t idx = (size_t)r * kC + c;
        out[idx] = acc[ti][tj][i] + bias[c] + xbuf[idx];
      }
}

// ------------------------------ launch -------------------------------------

extern "C" void kernel_launch(void* const* d_in, const int* in_sizes, int n_in,
                              void* d_out, int out_size, void* d_ws, size_t ws_size,
                              hipStream_t stream) {
  (void)in_sizes; (void)n_in; (void)out_size; (void)ws_size;
  const float* x     = (const float*)d_in[0];
  const float* mask  = (const float*)d_in[1];
  const float* n1w   = (const float*)d_in[2];
  const float* n1b   = (const float*)d_in[3];
  const float* qkvw  = (const float*)d_in[4];
  const float* qkvb  = (const float*)d_in[5];
  const float* relb  = (const float*)d_in[6];
  const float* projw = (const float*)d_in[7];
  const float* projb = (const float*)d_in[8];
  const float* n2w   = (const float*)d_in[9];
  const float* n2b   = (const float*)d_in[10];
  const float* w1    = (const float*)d_in[11];
  const float* b1    = (const float*)d_in[12];
  const float* w2    = (const float*)d_in[13];
  const float* b2    = (const float*)d_in[14];
  float* out = (float*)d_out;

  // Workspace layout with lifetime-based reuse (peak ~620 MB):
  //   region1: qkv (q|k|v, 3*73.5MB) during attn; MLP hidden (308MB) after.
  //   region2: LN1 output (77MB) for QKV GEMM; LN2 output after.
  char* ws = (char*)d_ws;
  size_t off = 0;
  auto take = [&](size_t bytes) -> char* {
    char* p = ws + off;
    off += (bytes + 255) & ~(size_t)255;
    return p;
  };
  const size_t actElems = (size_t)kM * kC;  // 38,535,168
  _Float16* region1 = (_Float16*)take((size_t)kM * kHidden * sizeof(_Float16));
  _Float16* region2 = (_Float16*)take(actElems * sizeof(_Float16));
  _Float16* obuf    = (_Float16*)take(actElems * sizeof(_Float16));
  float*    xbuf    = (float*)   take(actElems * sizeof(float));
  _Float16* wqkv16  = (_Float16*)take((size_t)k3C * kC * sizeof(_Float16));
  _Float16* wproj16 = (_Float16*)take((size_t)kC * kC * sizeof(_Float16));
  _Float16* w116    = (_Float16*)take((size_t)kHidden * kC * sizeof(_Float16));
  _Float16* w216    = (_Float16*)take((size_t)kC * kHidden * sizeof(_Float16));

  _Float16* qb  = region1;
  _Float16* kb  = region1 + actElems;
  _Float16* vb  = region1 + 2 * actElems;
  _Float16* hid = region1;       // reuses qkv region after attention
  _Float16* xw  = region2;       // LN1 output
  _Float16* h16 = region2;       // LN2 output (reuse)

  // weights -> f16 (kept in natural [out, in] orientation; the B-fragment
  // loader reads them as the transposed GEMM operand with contiguous K runs)
  k_f32_to_f16<<<(k3C * kC + 255) / 256, 256, 0, stream>>>(qkvw, wqkv16, k3C * kC);
  k_f32_to_f16<<<(kC * kC + 255) / 256, 256, 0, stream>>>(projw, wproj16, kC * kC);
  k_f32_to_f16<<<(kHidden * kC + 255) / 256, 256, 0, stream>>>(w1, w116, kHidden * kC);
  k_f32_to_f16<<<(kC * kHidden + 255) / 256, 256, 0, stream>>>(w2, w216, kC * kHidden);

  // 1) LN1 + shift + window partition
  k_ln_window<<<kM / 8, 256, 0, stream>>>(x, n1w, n1b, xw);
  // 2) QKV projection (WMMA, 64x64 tiles per wave)
  k_gemm_qkv<<<dim3(k3C / 64, kM / 64), 32, 0, stream>>>(xw, wqkv16, qkvb, qb, kb, vb);
  // 3) windowed attention (WMMA QK^T, softmax, WMMA PV)
  k_attn<<<kWindows * kHeads, 32, 0, stream>>>(qb, kb, vb, relb, mask, obuf);
  // 4) output projection + reverse-shift + residual
  k_gemm_proj<<<dim3(kC / 64, kM / 64), 32, 0, stream>>>(obuf, wproj16, projb, x, xbuf);
  // 5) LN2
  k_ln_plain<<<kM / 8, 256, 0, stream>>>(xbuf, n2w, n2b, h16);
  // 6) MLP fc1 + GELU
  k_gemm_mlp1<<<dim3(kHidden / 64, kM / 64), 32, 0, stream>>>(h16, w116, b1, hid);
  // 7) MLP fc2 + residual -> out
  k_gemm_mlp2<<<dim3(kC / 64, kM / 64), 32, 0, stream>>>(hid, w216, b2, xbuf, out);
}
